// GCNIDS_29480655519935
// MI455X (gfx1250) — matrix operand order repaired and checked by reading, not compiled, not run on hardware
//
#include <hip/hip_runtime.h>

typedef __attribute__((ext_vector_type(16))) _Float16 v16h;
typedef __attribute__((ext_vector_type(8)))  _Float16 v8h;
typedef __attribute__((ext_vector_type(8)))  float    v8f;

#define BN_EPS 1e-5f

// ---------------- small utility kernels ----------------

__global__ void zero_f(float* __restrict__ p, int n) {
    int i = blockIdx.x * blockDim.x + threadIdx.x;
    if (i < n) p[i] = 0.0f;
}

__global__ void deg_count(const int* __restrict__ dst, float* __restrict__ deg, int E) {
    int e = blockIdx.x * blockDim.x + threadIdx.x;
    if (e < E) atomicAdd(&deg[dst[e]], 1.0f);
}

__global__ void dis_fin(float* __restrict__ d, int n) {
    int i = blockIdx.x * blockDim.x + threadIdx.x;
    if (i < n) d[i] = rsqrtf(d[i] + 1.0f);
}

// W [K][Nout] fp32 row-major  ->  Wt [Nout][K] f16 (transposed for WMMA B-frag)
__global__ void convert_wt(const float* __restrict__ W, _Float16* __restrict__ Wt,
                           int K, int Nout, int lgN) {
    int idx = blockIdx.x * blockDim.x + threadIdx.x;
    if (idx >= K * Nout) return;
    int k = idx >> lgN;
    int n = idx & (Nout - 1);
    Wt[(size_t)n * K + k] = (_Float16)W[idx];
}

__global__ void f2h(const float* __restrict__ x, _Float16* __restrict__ y, int total) {
    int i = blockIdx.x * blockDim.x + threadIdx.x;
    if (i < total) y[i] = (_Float16)x[i];
}

// ---------------- WMMA GEMM with fused self-loop epilogue ----------------
// C[M x Nout]   = A16[M x K] * Wt16[Nout x K]^T              (linear output h)
// agg[M x Nout] = C * dis[row]^2 + bias[col]                 (self-loop + bias init)
// One wave computes a 16x64 tile (4 accumulators). blockDim = (32,4).
// A-frag (f16 16x32): lane = hi*16+lr ; M = m0+lr ; elems j=0..15 -> K = k+hi*16+j
//                     (contiguous, matching B-frag)
// B-frag (f16 32x16): N = n+lr ; elems j=0..15 -> K = k+hi*16+j  (contiguous in Wt row)
// C/D (f32 16x16): elem r -> row m0 + r + hi*8, col n + lr
__global__ void __launch_bounds__(128)
gemm_wmma_f16(const _Float16* __restrict__ A, const _Float16* __restrict__ Bt,
              float* __restrict__ C, const float* __restrict__ dis,
              const float* __restrict__ bias, float* __restrict__ agg,
              int K, int Nout) {
    const int lane = threadIdx.x;
    const int hi = lane >> 4, lr = lane & 15;
    const int m0 = blockIdx.x * 16;
    const int n0 = (blockIdx.y * blockDim.y + threadIdx.y) * 64;

    const _Float16* arow = A + (size_t)(m0 + lr) * K;
    v8f acc[4] = {};

    for (int k = 0; k < K; k += 32) {
        v16h a = *(const v16h*)(arow + k + hi * 16);
#pragma unroll
        for (int t = 0; t < 4; ++t) {
            const _Float16* bp = Bt + (size_t)(n0 + t * 16 + lr) * K + k + hi * 16;
            v16h b = *(const v16h*)bp;
            acc[t] = __builtin_amdgcn_wmma_f32_16x16x32_f16(
                false, a, false, b, (short)0, acc[t], false, false);
        }
    }

    // per-row self-loop weights (8 rows owned by this lane)
    float dsq[8];
#pragma unroll
    for (int r = 0; r < 8; ++r) {
        float dd = dis[m0 + r + hi * 8];
        dsq[r] = dd * dd;
    }
#pragma unroll
    for (int t = 0; t < 4; ++t) {
        const int col = n0 + t * 16 + lr;
        const float bcol = bias[col];
#pragma unroll
        for (int r = 0; r < 8; ++r) {
            const size_t idx = (size_t)(m0 + r + hi * 8) * Nout + col;
            const float v = acc[t][r];
            C[idx]   = v;
            agg[idx] = v * dsq[r] + bcol;
        }
    }
}

// ---------------- graph aggregation ----------------
// blockDim = (F/4, 2): each y-slice handles one edge, float4 gather + 4 atomics.
// agg[dst] += h[src] * dis[src]*dis[dst]
__global__ void edge_scatter(const float* __restrict__ h, const int* __restrict__ src,
                             const int* __restrict__ dst, const float* __restrict__ dis,
                             float* __restrict__ agg, int F, int E) {
    int e = blockIdx.x * blockDim.y + threadIdx.y;
    if (e >= E) return;

    // prefetch a look-ahead edge's source row (global_prefetch_b8) to hide
    // the dependent src[e] -> h[src] gather latency
    int e2 = e + 4 * (int)blockDim.y;
    if (e2 < E && threadIdx.x < (unsigned)(F / 32)) {
        const float* prow = h + (size_t)src[e2] * F + threadIdx.x * 32;
        __builtin_prefetch(prow, 0, 1);
    }

    int s = src[e], d = dst[e];
    float norm = dis[s] * dis[d];
    const float4* hs = (const float4*)(h + (size_t)s * F);
    float* ad = agg + (size_t)d * F;

    float4 v = hs[threadIdx.x];
    int f4 = threadIdx.x * 4;
    atomicAdd(&ad[f4 + 0], v.x * norm);
    atomicAdd(&ad[f4 + 1], v.y * norm);
    atomicAdd(&ad[f4 + 2], v.z * norm);
    atomicAdd(&ad[f4 + 3], v.w * norm);
}

// ---------------- batchnorm + relu ----------------

// blockDim.x = F; each block reduces a chunk of rows for all F columns (coalesced)
__global__ void bn_stats(const float* __restrict__ x, float* __restrict__ sum,
                         float* __restrict__ sumsq, int N, int F, int rpb) {
    int f = threadIdx.x;
    int r0 = blockIdx.x * rpb;
    int r1 = r0 + rpb; if (r1 > N) r1 = N;
    float s = 0.f, s2 = 0.f;
    for (int r = r0; r < r1; ++r) {
        float v = x[(size_t)r * F + f];
        s += v; s2 += v * v;
    }
    atomicAdd(&sum[f], s);
    atomicAdd(&sumsq[f], s2);
}

__global__ void bn_finalize(const float* __restrict__ sum, const float* __restrict__ sumsq,
                            const float* __restrict__ g, const float* __restrict__ be,
                            float* __restrict__ sc, float* __restrict__ sh, int N, int F) {
    int f = blockIdx.x * blockDim.x + threadIdx.x;
    if (f >= F) return;
    float inv = 1.0f / (float)N;
    float mu = sum[f] * inv;
    float var = sumsq[f] * inv - mu * mu;
    float s = g[f] * rsqrtf(var + BN_EPS);
    sc[f] = s;
    sh[f] = be[f] - mu * s;
}

// y = relu(x*sc + sh); optionally also write f16 copy for the next WMMA GEMM
__global__ void bn_apply(const float* __restrict__ x, const float* __restrict__ sc,
                         const float* __restrict__ sh, float* __restrict__ y,
                         _Float16* __restrict__ y16, int total, int fmask) {
    int i = blockIdx.x * blockDim.x + threadIdx.x;
    if (i >= total) return;
    int f = i & fmask;
    float v = x[i] * sc[f] + sh[f];
    v = v > 0.f ? v : 0.f;
    y[i] = v;
    if (y16) y16[i] = (_Float16)v;
}

// ---------------- final 256 -> 10 projection (wave-shuffle reduction) ----------------
__global__ void final_gemm(const float* __restrict__ h, const float* __restrict__ Wout,
                           const float* __restrict__ bout, float* __restrict__ out, int C) {
    int row = blockIdx.x, lane = threadIdx.x;
    const float* hr = h + (size_t)row * 256;
    float hv[8];
#pragma unroll
    for (int j = 0; j < 8; ++j) hv[j] = hr[lane + 32 * j];
    for (int c = 0; c < C; ++c) {
        float s = 0.f;
#pragma unroll
        for (int j = 0; j < 8; ++j) s += hv[j] * Wout[(lane + 32 * j) * C + c];
#pragma unroll
        for (int o = 16; o >= 1; o >>= 1) s += __shfl_xor(s, o, 32);
        if (lane == 0) out[(size_t)row * C + c] = s + bout[c];
    }
}

// ---------------- host launcher ----------------

extern "C" void kernel_launch(void* const* d_in, const int* in_sizes, int n_in,
                              void* d_out, int out_size, void* d_ws, size_t ws_size,
                              hipStream_t stream) {
    const float* x    = (const float*)d_in[0];
    const int*   ei   = (const int*)  d_in[1];
    const float* W1   = (const float*)d_in[2];
    const float* b1   = (const float*)d_in[3];
    const float* g1   = (const float*)d_in[4];
    const float* be1  = (const float*)d_in[5];
    const float* W2   = (const float*)d_in[6];
    const float* b2   = (const float*)d_in[7];
    const float* g2   = (const float*)d_in[8];
    const float* be2  = (const float*)d_in[9];
    const float* W3   = (const float*)d_in[10];
    const float* b3   = (const float*)d_in[11];
    const float* g3   = (const float*)d_in[12];
    const float* be3  = (const float*)d_in[13];
    const float* Wout = (const float*)d_in[14];
    const float* bout = (const float*)d_in[15];
    float* out = (float*)d_out;

    const int D = 512, H = 512, O = 256, C = 10;
    const int N = in_sizes[0] / D;        // 10000
    const int E = in_sizes[1] / 2;        // 160000
    const int* src = ei;
    const int* dst = ei + E;

    // workspace carve-out (256B aligned)
    char* ws = (char*)d_ws;
    size_t off = 0;
    auto alloc = [&](size_t bytes) -> void* {
        size_t o = (off + 255) & ~(size_t)255;
        off = o + bytes;
        return (void*)(ws + o);
    };
    float*    dis   = (float*)   alloc((size_t)N * 4);
    float*    sum   = (float*)   alloc(512 * 4);
    float*    sumsq = (float*)   alloc(512 * 4);
    float*    sc    = (float*)   alloc(512 * 4);
    float*    sh    = (float*)   alloc(512 * 4);
    _Float16* w1t   = (_Float16*)alloc((size_t)D * H * 2);
    _Float16* w2t   = (_Float16*)alloc((size_t)H * H * 2);
    _Float16* w3t   = (_Float16*)alloc((size_t)H * O * 2);
    _Float16* h16   = (_Float16*)alloc((size_t)N * H * 2);
    float*    bufA  = (float*)   alloc((size_t)N * H * 4);
    float*    bufB  = (float*)   alloc((size_t)N * H * 4);

    const int T = 256;
    const int rpb = 125;                       // rows per bn_stats block
    const int bnBlocks = (N + rpb - 1) / rpb;
    dim3 gwave(32, 4);
    dim3 scat512(128, 2), scat256(64, 2);      // (F/4 threads per edge, 2 edges/block)
    const int escBlocks = (E + 1) / 2;

    // degrees -> dis = rsqrt(deg+1)
    zero_f<<<(N + T - 1) / T, T, 0, stream>>>(dis, N);
    deg_count<<<(E + T - 1) / T, T, 0, stream>>>(dst, dis, E);
    dis_fin<<<(N + T - 1) / T, T, 0, stream>>>(dis, N);

    // transpose+convert weights to f16
    convert_wt<<<(D * H + T - 1) / T, T, 0, stream>>>(W1, w1t, D, H, 9);
    convert_wt<<<(H * H + T - 1) / T, T, 0, stream>>>(W2, w2t, H, H, 9);
    convert_wt<<<(H * O + T - 1) / T, T, 0, stream>>>(W3, w3t, H, O, 8);

    // ---- layer 1 ----
    f2h<<<(N * D + T - 1) / T, T, 0, stream>>>(x, h16, N * D);
    gemm_wmma_f16<<<dim3(N / 16, H / 256), gwave, 0, stream>>>(h16, w1t, bufA,
                                                               dis, b1, bufB, D, H);
    edge_scatter<<<escBlocks, scat512, 0, stream>>>(bufA, src, dst, dis, bufB, H, E);
    zero_f<<<(H + T - 1) / T, T, 0, stream>>>(sum, H);
    zero_f<<<(H + T - 1) / T, T, 0, stream>>>(sumsq, H);
    bn_stats<<<bnBlocks, H, 0, stream>>>(bufB, sum, sumsq, N, H, rpb);
    bn_finalize<<<(H + T - 1) / T, T, 0, stream>>>(sum, sumsq, g1, be1, sc, sh, N, H);
    bn_apply<<<(N * H + T - 1) / T, T, 0, stream>>>(bufB, sc, sh, bufA, h16, N * H, H - 1);

    // ---- layer 2 ----
    gemm_wmma_f16<<<dim3(N / 16, H / 256), gwave, 0, stream>>>(h16, w2t, bufB,
                                                               dis, b2, bufA, H, H);
    edge_scatter<<<escBlocks, scat512, 0, stream>>>(bufB, src, dst, dis, bufA, H, E);
    zero_f<<<(H + T - 1) / T, T, 0, stream>>>(sum, H);
    zero_f<<<(H + T - 1) / T, T, 0, stream>>>(sumsq, H);
    bn_stats<<<bnBlocks, H, 0, stream>>>(bufA, sum, sumsq, N, H, rpb);
    bn_finalize<<<(H + T - 1) / T, T, 0, stream>>>(sum, sumsq, g2, be2, sc, sh, N, H);
    bn_apply<<<(N * H + T - 1) / T, T, 0, stream>>>(bufA, sc, sh, bufB, h16, N * H, H - 1);

    // ---- layer 3 (output width 256) ----
    gemm_wmma_f16<<<dim3(N / 16, O / 256), gwave, 0, stream>>>(h16, w3t, bufA,
                                                               dis, b3, bufB, H, O);
    edge_scatter<<<escBlocks, scat256, 0, stream>>>(bufA, src, dst, dis, bufB, O, E);
    zero_f<<<(O + T - 1) / T, T, 0, stream>>>(sum, O);
    zero_f<<<(O + T - 1) / T, T, 0, stream>>>(sumsq, O);
    bn_stats<<<bnBlocks, O, 0, stream>>>(bufB, sum, sumsq, N, O, rpb);
    bn_finalize<<<(O + T - 1) / T, T, 0, stream>>>(sum, sumsq, g3, be3, sc, sh, N, O);
    bn_apply<<<(N * O + T - 1) / T, T, 0, stream>>>(bufB, sc, sh, bufA,
                                                    (_Float16*)nullptr, N * O, O - 1);

    // ---- output projection ----
    final_gemm<<<N, 32, 0, stream>>>(bufA, Wout, bout, out, C);
}
